// BondConv_17437567222208
// MI455X (gfx1250) — compile-verified, matrix operand order checked by reading, not run on hardware
//
#include <hip/hip_runtime.h>

// ---------------------------------------------------------------------------
// CDNA5 (gfx1250) fused bond-convolution.
// Memory-bound (~300MB HBM traffic vs 43 GFLOP). Design:
//  - fuse all gathers (never materialize the 512MB "total" matrix)
//  - all MLP weights bf16-transposed in LDS (80KB of the 320KB WGP LDS)
//  - v_wmma_f32_16x16x32_bf16 with f32 accumulation
//  - M=32 per wave: each LDS B-fragment feeds two A-tiles (halves LDS traffic)
//  - B-fragments double-buffered in pairs to avoid WMMA WAR-hazard v_nops
//  - non-temporal loads on single-use streams (angle_feas / agg) so the
//    gathered tables (~112MB) stay resident in the 192MB L2
//  - segment-sum via global_atomic_add_f32 into d_out
// ---------------------------------------------------------------------------

typedef __attribute__((ext_vector_type(16))) __bf16 v16bf;
typedef __attribute__((ext_vector_type(8)))  __bf16 v8bf;
typedef __attribute__((ext_vector_type(8)))  float  v8f;
typedef __attribute__((ext_vector_type(4)))  float  v4f;

#define NA_   40000
#define NB_   200000
#define NG_   500000
#define D_    64
#define IN_   256
#define WV    8          // waves per block (256 threads)

__device__ __forceinline__ float fsig(float x) { return 1.0f / (1.0f + __expf(-x)); }
__device__ __forceinline__ float fsilu(float x){ return x * fsig(x); }

__device__ __forceinline__ v8f wmma_bf16(v16bf a, v16bf b, v8f c) {
  return __builtin_amdgcn_wmma_f32_16x16x32_bf16(false, a, false, b, (short)0, c,
                                                 false, false);
}

// A-fragment (16x32 bf16) from a contiguous f32 row segment (regular loads;
// these hit the L2-resident gather tables).
__device__ __forceinline__ v16bf a_from_f32(const float* p) {
  v16bf r;
#pragma unroll
  for (int i = 0; i < 8; ++i) {
    r[i]     = (__bf16)p[i];
    r[i + 8] = (__bf16)p[i + 16];
  }
  return r;
}

// Same but with non-temporal loads: used for streams read exactly once so
// they do not evict the gather tables from L2.
__device__ __forceinline__ v16bf a_from_f32_nt(const float* p) {
  v4f x0 = __builtin_nontemporal_load((const v4f*)(p));
  v4f x1 = __builtin_nontemporal_load((const v4f*)(p + 4));
  v4f x2 = __builtin_nontemporal_load((const v4f*)(p + 16));
  v4f x3 = __builtin_nontemporal_load((const v4f*)(p + 20));
  v16bf r;
#pragma unroll
  for (int i = 0; i < 4; ++i) {
    r[i]      = (__bf16)x0[i];
    r[i + 4]  = (__bf16)x1[i];
    r[i + 8]  = (__bf16)x2[i];
    r[i + 12] = (__bf16)x3[i];
  }
  return r;
}

// Fragment from LDS bf16 (two 16B chunks 16 elements apart).
__device__ __forceinline__ v16bf frag_lds(const __bf16* p) {
  v8bf lo = *(const v8bf*)(p);
  v8bf hi = *(const v8bf*)(p + 16);
  v16bf r;
#pragma unroll
  for (int i = 0; i < 8; ++i) { r[i] = lo[i]; r[i + 8] = hi[i]; }
  return r;
}

// ---------------------------------------------------------------------------
__global__ void zero_f4_kernel(v4f* p, long n4) {
  long i = (long)blockIdx.x * blockDim.x + threadIdx.x;
  long stride = (long)gridDim.x * blockDim.x;
  v4f z = {0.f, 0.f, 0.f, 0.f};
  for (; i < n4; i += stride) p[i] = z;
}

// ---------------------------------------------------------------------------
// Kernel 1: per-angle fused MLP + gated update + atomic segment-sum into agg.
// One wave per 32-angle tile (two 16-row WMMA M-tiles share every B-fragment).
// ---------------------------------------------------------------------------
__global__ void __launch_bounds__(32 * WV)
angle_kernel(const float* __restrict__ atom_feas,
             const float* __restrict__ bond_feas,
             const float* __restrict__ bond_weights,
             const float* __restrict__ angle_feas,
             const int*   __restrict__ bgph,
             const float* __restrict__ Wc1, const float* __restrict__ bc1,
             const float* __restrict__ Wc2, const float* __restrict__ bc2,
             const float* __restrict__ Wg1, const float* __restrict__ bg1,
             const float* __restrict__ Wg2, const float* __restrict__ bg2,
             float* agg) {
  // Weights transposed to [N][K] in LDS so B fragments are contiguous b128s.
  __shared__ __align__(16) __bf16 s_w1t[2][64 * 256];   // Wc1^T, Wg1^T  (64KB)
  __shared__ __align__(16) __bf16 s_w2t[2][64 * 64];    // Wc2^T, Wg2^T  (16KB)
  __shared__ float s_bias[4][64];                       // bc1,bg1,bc2,bg2
  __shared__ int   s_idx[WV][2][32];                    // bi, bj per tile row
  __shared__ __align__(16) __bf16 s_hid[WV][2][32 * 64]; // hidden C/G (64KB)

  for (int i = threadIdx.x; i < 256 * 64; i += blockDim.x) {
    int k = i >> 6, n = i & 63;
    s_w1t[0][n * 256 + k] = (__bf16)Wc1[i];
    s_w1t[1][n * 256 + k] = (__bf16)Wg1[i];
  }
  for (int i = threadIdx.x; i < 64 * 64; i += blockDim.x) {
    int k = i >> 6, n = i & 63;
    s_w2t[0][n * 64 + k] = (__bf16)Wc2[i];
    s_w2t[1][n * 64 + k] = (__bf16)Wg2[i];
  }
  if (threadIdx.x < 64) {
    int n = threadIdx.x;
    s_bias[0][n] = bc1[n]; s_bias[1][n] = bg1[n];
    s_bias[2][n] = bc2[n]; s_bias[3][n] = bg2[n];
  }
  __syncthreads();

  const int wave = threadIdx.x >> 5;
  const int lane = threadIdx.x & 31;
  const int row  = lane & 15;       // M row within each A-tile / N col of B
  const int sub  = lane >> 4;       // which K-half this lane carries
  const int tile = blockIdx.x * WV + wave;
  const int NT   = NG_ / 32;        // 15625 exact
  if (tile >= NT) return;

  // Two M-tiles: rows g0 and g1 of this lane.
  const long g0 = (long)tile * 32 + row;
  const long g1 = g0 + 16;
  const int ci0 = bgph[g0 * 3 + 0], bi0 = bgph[g0 * 3 + 1], bj0 = bgph[g0 * 3 + 2];
  const int ci1 = bgph[g1 * 3 + 0], bi1 = bgph[g1 * 3 + 1], bj1 = bgph[g1 * 3 + 2];
  if (lane < 16) {
    s_idx[wave][0][row]      = bi0;  s_idx[wave][1][row]      = bj0;
    s_idx[wave][0][row + 16] = bi1;  s_idx[wave][1][row + 16] = bj1;
  }

  // Virtual 256-wide input rows, as four 64-wide gathered segments each.
  const float* segA[4] = { bond_feas + (long)bi0 * 64, bond_feas + (long)bj0 * 64,
                           angle_feas + g0 * 64,       atom_feas + (long)ci0 * 64 };
  const float* segB[4] = { bond_feas + (long)bi1 * 64, bond_feas + (long)bj1 * 64,
                           angle_feas + g1 * 64,       atom_feas + (long)ci1 * 64 };

  // ---- GEMM1: (32x256) @ (256x64), core + gate paths ----------------------
  v8f accC[2][4] = {}, accG[2][4] = {};
#pragma unroll
  for (int k = 0; k < 8; ++k) {                 // K-chunks of 32
    const int s  = k >> 1;
    const int co = (k & 1) * 32 + sub * 8;
    const int kb = k * 32 + sub * 8;
    v16bf a0 = (s == 2) ? a_from_f32_nt(segA[2] + co) : a_from_f32(segA[s] + co);
    v16bf a1 = (s == 2) ? a_from_f32_nt(segB[2] + co) : a_from_f32(segB[s] + co);
#pragma unroll
    for (int np = 0; np < 2; ++np) {            // nt-pairs: 4 live B fragments
      const int n0 = np * 2, n1 = np * 2 + 1;
      v16bf bC0 = frag_lds(&s_w1t[0][(n0 * 16 + row) * 256 + kb]);
      v16bf bG0 = frag_lds(&s_w1t[1][(n0 * 16 + row) * 256 + kb]);
      v16bf bC1 = frag_lds(&s_w1t[0][(n1 * 16 + row) * 256 + kb]);
      v16bf bG1 = frag_lds(&s_w1t[1][(n1 * 16 + row) * 256 + kb]);
      accC[0][n0] = wmma_bf16(a0, bC0, accC[0][n0]);
      accC[1][n0] = wmma_bf16(a1, bC0, accC[1][n0]);
      accG[0][n0] = wmma_bf16(a0, bG0, accG[0][n0]);
      accG[1][n0] = wmma_bf16(a1, bG0, accG[1][n0]);
      accC[0][n1] = wmma_bf16(a0, bC1, accC[0][n1]);
      accC[1][n1] = wmma_bf16(a1, bC1, accC[1][n1]);
      accG[0][n1] = wmma_bf16(a0, bG1, accG[0][n1]);
      accG[1][n1] = wmma_bf16(a1, bG1, accG[1][n1]);
    }
  }

  // ---- bias + silu, write hidden (32x64) to per-wave LDS as bf16 ----------
  __bf16* hc = &s_hid[wave][0][0];
  __bf16* hg = &s_hid[wave][1][0];
#pragma unroll
  for (int mt = 0; mt < 2; ++mt) {
#pragma unroll
    for (int nt = 0; nt < 4; ++nt) {
      const int n  = nt * 16 + row;             // hidden unit (C-layout col)
      const float b1 = s_bias[0][n];
      const float b2 = s_bias[1][n];
#pragma unroll
      for (int j = 0; j < 8; ++j) {
        const int m = mt * 16 + j + sub * 8;    // tile row (C-layout)
        hc[m * 64 + n] = (__bf16)fsilu(accC[mt][nt][j] + b1);
        hg[m * 64 + n] = (__bf16)fsilu(accG[mt][nt][j] + b2);
      }
    }
  }
  // In-wave LDS RAW: compiler inserts s_wait_dscnt; wave32 lockstep = sync.

  // ---- GEMM2: (32x64) @ (64x64) -------------------------------------------
  v8f acc2C[2][4] = {}, acc2G[2][4] = {};
#pragma unroll
  for (int k = 0; k < 2; ++k) {
    const int c0 = k * 32 + sub * 8;
    v16bf aC0 = frag_lds(&hc[(row)      * 64 + c0]);
    v16bf aC1 = frag_lds(&hc[(row + 16) * 64 + c0]);
    v16bf aG0 = frag_lds(&hg[(row)      * 64 + c0]);
    v16bf aG1 = frag_lds(&hg[(row + 16) * 64 + c0]);
#pragma unroll
    for (int np = 0; np < 2; ++np) {
      const int n0 = np * 2, n1 = np * 2 + 1;
      v16bf bC0 = frag_lds(&s_w2t[0][(n0 * 16 + row) * 64 + c0]);
      v16bf bG0 = frag_lds(&s_w2t[1][(n0 * 16 + row) * 64 + c0]);
      v16bf bC1 = frag_lds(&s_w2t[0][(n1 * 16 + row) * 64 + c0]);
      v16bf bG1 = frag_lds(&s_w2t[1][(n1 * 16 + row) * 64 + c0]);
      acc2C[0][n0] = wmma_bf16(aC0, bC0, acc2C[0][n0]);
      acc2C[1][n0] = wmma_bf16(aC1, bC0, acc2C[1][n0]);
      acc2G[0][n0] = wmma_bf16(aG0, bG0, acc2G[0][n0]);
      acc2G[1][n0] = wmma_bf16(aG1, bG0, acc2G[1][n0]);
      acc2C[0][n1] = wmma_bf16(aC0, bC1, acc2C[0][n1]);
      acc2C[1][n1] = wmma_bf16(aC1, bC1, acc2C[1][n1]);
      acc2G[0][n1] = wmma_bf16(aG0, bG1, acc2G[0][n1]);
      acc2G[1][n1] = wmma_bf16(aG1, bG1, acc2G[1][n1]);
    }
  }

  // ---- epilogue: silu * sigmoid * bw[bi] * bw[bj], atomic segment-sum -----
#pragma unroll
  for (int mt = 0; mt < 2; ++mt) {
    int bim[8], bjm[8];
#pragma unroll
    for (int j = 0; j < 8; ++j) {
      const int m = mt * 16 + j + sub * 8;
      bim[j] = s_idx[wave][0][m];
      bjm[j] = s_idx[wave][1][m];
    }
#pragma unroll
    for (int nt = 0; nt < 4; ++nt) {
      const int n   = nt * 16 + row;
      const float b2c = s_bias[2][n];
      const float b2g = s_bias[3][n];
#pragma unroll
      for (int j = 0; j < 8; ++j) {
        const float core = fsilu(acc2C[mt][nt][j] + b2c);
        const float gate = fsig (acc2G[mt][nt][j] + b2g);
        const float w = bond_weights[(long)bim[j] * 64 + n] *
                        bond_weights[(long)bjm[j] * 64 + n];
        unsafeAtomicAdd(&agg[(long)bim[j] * 64 + n], core * gate * w);
      }
    }
  }
}

// ---------------------------------------------------------------------------
// Kernel 2: out = agg @ Wo + bo + bond_feas, in-place on d_out.
// agg rows are last-use -> non-temporal loads; out is write-once -> NT store.
// ---------------------------------------------------------------------------
__global__ void __launch_bounds__(32 * WV)
bond_out_kernel(const float* agg,
                const float* __restrict__ bond_feas,
                const float* __restrict__ Wo,
                const float* __restrict__ bo,
                float* out) {
  __shared__ __align__(16) __bf16 s_wot[64 * 64];
  __shared__ float s_bo[64];
  for (int i = threadIdx.x; i < 64 * 64; i += blockDim.x) {
    int k = i >> 6, n = i & 63;
    s_wot[n * 64 + k] = (__bf16)Wo[i];
  }
  if (threadIdx.x < 64) s_bo[threadIdx.x] = bo[threadIdx.x];
  __syncthreads();

  const int wave = threadIdx.x >> 5;
  const int lane = threadIdx.x & 31;
  const int row  = lane & 15;
  const int sub  = lane >> 4;
  const int tile = blockIdx.x * WV + wave;
  const int NT   = NB_ / 16;        // 12500 exact
  if (tile >= NT) return;

  const long b0 = (long)tile * 16;
  v8f acc[4] = {};
#pragma unroll
  for (int k = 0; k < 2; ++k) {
    const float* p = agg + (b0 + row) * 64 + k * 32 + sub * 8;
    v16bf a = a_from_f32_nt(p);
#pragma unroll
    for (int nt = 0; nt < 4; ++nt) {
      v16bf bf = frag_lds(&s_wot[(nt * 16 + row) * 64 + k * 32 + sub * 8]);
      acc[nt] = wmma_bf16(a, bf, acc[nt]);
    }
  }
#pragma unroll
  for (int nt = 0; nt < 4; ++nt) {
    const int n = nt * 16 + row;
    const float bb = s_bo[n];
#pragma unroll
    for (int j = 0; j < 8; ++j) {
      const int m = j + sub * 8;
      const long o = (b0 + m) * 64 + n;
      __builtin_nontemporal_store(acc[nt][j] + bb + bond_feas[o], &out[o]);
    }
  }
}

// ---------------------------------------------------------------------------
extern "C" void kernel_launch(void* const* d_in, const int* in_sizes, int n_in,
                              void* d_out, int out_size, void* d_ws, size_t ws_size,
                              hipStream_t stream) {
  const float* atom_feas    = (const float*)d_in[0];
  const float* bond_feas    = (const float*)d_in[1];
  const float* bond_weights = (const float*)d_in[2];
  const float* angle_feas   = (const float*)d_in[3];
  const int*   bond_graph   = (const int*)  d_in[4];
  const float* Wc1 = (const float*)d_in[5];
  const float* bc1 = (const float*)d_in[6];
  const float* Wc2 = (const float*)d_in[7];
  const float* bc2 = (const float*)d_in[8];
  const float* Wg1 = (const float*)d_in[9];
  const float* bg1 = (const float*)d_in[10];
  const float* Wg2 = (const float*)d_in[11];
  const float* bg2 = (const float*)d_in[12];
  const float* Wo  = (const float*)d_in[13];
  const float* bo  = (const float*)d_in[14];
  float* out = (float*)d_out;

  // 1) zero the aggregation buffer (d_out doubles as agg; poisoned by harness).
  const long n4 = (long)NB_ * D_ / 4;
  zero_f4_kernel<<<4096, 256, 0, stream>>>((v4f*)out, n4);

  // 2) angle pass: fused gather + 2x MLP + gated update + atomic segment-sum.
  const int nt1 = NG_ / 32;
  angle_kernel<<<(nt1 + WV - 1) / WV, 32 * WV, 0, stream>>>(
      atom_feas, bond_feas, bond_weights, angle_feas, bond_graph,
      Wc1, bc1, Wc2, bc2, Wg1, bg1, Wg2, bg2, out);

  // 3) bond pass: out = agg @ Wo + bo + bond_feas (in place).
  const int nt2 = NB_ / 16;
  bond_out_kernel<<<(nt2 + WV - 1) / WV, 32 * WV, 0, stream>>>(
      out, bond_feas, Wo, bo, out);
}
